// ScannedRNN_47218870452552
// MI455X (gfx1250) — matrix operand order repaired and checked by reading, not compiled, not run on hardware
//
#include <hip/hip_runtime.h>
#include <hip/hip_bf16.h>
#include <stddef.h>

// ---------------- problem constants ----------------
#define T_STEPS 512
#define BATCH   64
#define DH      1024
#define G3      3
#define COLS    16                 // output columns per block per gate
#define NBLK    (DH / COLS)        // 64 blocks
#define NTHR    128                // 4 waves of 32

// fragment-interleaved LDS layout:
//  block(c, kc, e) = 6 frags x 32B = 192B, e in {0,1} half-wave
//  per-column stride padded to 12304B -> dword stride 3076 = 4 (mod 64)
#define FBLK    192
#define CSTRIDE 12304              // 32 kchunks * 2 e * 192 + 16 pad
#define SMEMSZ  (16 * CSTRIDE)     // 196,864 B

typedef __attribute__((ext_vector_type(16))) __bf16 bf16x16;
typedef __attribute__((ext_vector_type(8)))  float  f32x8;

union FragAB { unsigned u[8]; bf16x16 v; };

// ---------------- helpers ----------------
__device__ __forceinline__ unsigned pk_bf16(float a, float b) {
  unsigned ua = __builtin_bit_cast(unsigned, a);
  unsigned ub = __builtin_bit_cast(unsigned, b);
  ua = (ua + 0x7FFFu + ((ua >> 16) & 1u)) >> 16;   // round-to-nearest-even
  ub = (ub + 0x7FFFu + ((ub >> 16) & 1u)) >> 16;
  return (ua & 0xFFFFu) | (ub << 16);
}

__device__ __forceinline__ unsigned short f2bf(float f) {
  unsigned u = __builtin_bit_cast(unsigned, f);
  u = (u + 0x7FFFu + ((u >> 16) & 1u)) >> 16;
  return (unsigned short)u;
}

__device__ __forceinline__ f32x8 wmma_bf16(bf16x16 a, bf16x16 b, f32x8 c) {
  return __builtin_amdgcn_wmma_f32_16x16x32_bf16(false, a, false, b, (short)0, c,
                                                 false, false);
}

__device__ __forceinline__ float fsig(float v) {
  float e = __expf(-v);
  return __builtin_amdgcn_rcpf(1.0f + e);
}

__device__ __forceinline__ float ftanh(float v) {
  float e = __expf(2.0f * v);
  return 1.0f - 2.0f * __builtin_amdgcn_rcpf(e + 1.0f);
}

__device__ __forceinline__ void grid_barrier(unsigned* count, unsigned* gen,
                                             unsigned target) {
  __threadfence();
  __syncthreads();
  if (threadIdx.x == 0) {
    unsigned prev = __hip_atomic_fetch_add(count, 1u, __ATOMIC_ACQ_REL,
                                           __HIP_MEMORY_SCOPE_AGENT);
    if (prev == (unsigned)(NBLK - 1)) {
      __hip_atomic_store(count, 0u, __ATOMIC_RELAXED, __HIP_MEMORY_SCOPE_AGENT);
      __hip_atomic_store(gen, target, __ATOMIC_RELEASE, __HIP_MEMORY_SCOPE_AGENT);
    } else {
      while (__hip_atomic_load(gen, __ATOMIC_ACQUIRE, __HIP_MEMORY_SCOPE_AGENT) <
             target)
        __builtin_amdgcn_s_sleep(2);
    }
  }
  __syncthreads();
  __threadfence();
}

// ---------------- prep kernels ----------------
__global__ void prep_weights(const float* __restrict__ Wi,
                             const float* __restrict__ Wh,
                             unsigned short* __restrict__ wiT,
                             unsigned short* __restrict__ whT) {
  size_t i = (size_t)blockIdx.x * blockDim.x + threadIdx.x;
  const size_t N = (size_t)G3 * DH * DH;
  if (i >= N) return;
  size_t c = i / DH;
  size_t k = i % DH;
  wiT[i] = f2bf(Wi[k * (G3 * DH) + c]);
  whT[i] = f2bf(Wh[k * (G3 * DH) + c]);
}

__global__ void prep_x(const float* __restrict__ x, unsigned* __restrict__ xbf) {
  size_t i = (size_t)blockIdx.x * blockDim.x + threadIdx.x;
  const size_t N2 = (size_t)T_STEPS * BATCH * DH / 2;
  if (i >= N2) return;
  xbf[i] = pk_bf16(x[2 * i], x[2 * i + 1]);
}

__global__ void init_misc(unsigned* bar, unsigned* zbf_dw, float* zf32) {
  int i = threadIdx.x;                 // 1024 threads
  if (i < 2) bar[i] = 0u;
  if (i < DH / 2) zbf_dw[i] = 0u;
  zf32[i] = 0.0f;
}

// ---------------- persistent GRU scan ----------------
extern __shared__ unsigned char smem_raw[];

__device__ __forceinline__ void loadA(FragAB& a, const unsigned short* row, int k,
                                      int s) {
  uint4 a0 = *(const uint4*)(row + k + s);
  uint4 a1 = *(const uint4*)(row + k + 16 + s);
  a.u[0] = a0.x; a.u[1] = a0.y; a.u[2] = a0.z; a.u[3] = a0.w;
  a.u[4] = a1.x; a.u[5] = a1.y; a.u[6] = a1.z; a.u[7] = a1.w;
}

__device__ __forceinline__ void loadB6(FragAB* f, const unsigned char* p) {
#pragma unroll
  for (int q = 0; q < 6; ++q) {
    uint4 b0 = *(const uint4*)(p + q * 32);
    uint4 b1 = *(const uint4*)(p + q * 32 + 16);
    f[q].u[0] = b0.x; f[q].u[1] = b0.y; f[q].u[2] = b0.z; f[q].u[3] = b0.w;
    f[q].u[4] = b1.x; f[q].u[5] = b1.y; f[q].u[6] = b1.z; f[q].u[7] = b1.w;
  }
}

__device__ __forceinline__ void wmma6(const FragAB& ax, const FragAB& ah,
                                      const FragAB* f, f32x8* acc) {
#pragma unroll
  for (int g = G3 - 1; g >= 0; --g)
    acc[3 + g] = wmma_bf16(ah.v, f[3 + g].v, acc[3 + g]);
#pragma unroll
  for (int g = G3 - 1; g >= 0; --g)
    acc[g] = wmma_bf16(ax.v, f[g].v, acc[g]);
}

__global__ void __launch_bounds__(NTHR, 1)
gru_scan_kernel(const unsigned short* __restrict__ xbf,
                const float* __restrict__ bi, const float* __restrict__ bhn,
                const unsigned char* __restrict__ resets,
                const unsigned short* __restrict__ wiT,
                const unsigned short* __restrict__ whT,
                unsigned short* __restrict__ hbuf,       // 2 x [B][D] bf16
                const unsigned short* __restrict__ zbf,  // 1024 bf16 zeros
                const float* __restrict__ zf32,          // 1024 f32 zeros
                unsigned* __restrict__ bar, float* __restrict__ out) {
  const int tid      = threadIdx.x;
  const int wid      = tid >> 5;
  const int lane     = tid & 31;
  const int laneHigh = lane >> 4;
  const int colLocal = lane & 15;
  const int c0       = blockIdx.x * COLS;

  // ---- stage weights into fragment-interleaved LDS (once) ----
  // uint4 index space: (c:16) x (kc:32) x (e:2) x (f:6) x (half:2) = 12288
  for (int idx = tid; idx < 12288; idx += NTHR) {
    int half = idx & 1;
    int rest = idx >> 1;              // (c,kc,e,f)
    int f    = rest % 6;
    int e    = (rest / 6) & 1;
    int kc   = (rest / 12) & 31;
    int c    = rest / 384;
    int g    = (f < 3) ? f : f - 3;
    const unsigned short* W = (f < 3) ? wiT : whT;
    const uint4* src = (const uint4*)(W + ((size_t)g * DH + c0 + c) * DH +
                                      kc * 32 + half * 16 + e * 8);
    uint4* dst = (uint4*)(smem_raw + (size_t)c * CSTRIDE +
                          (size_t)(kc * 2 + e) * FBLK + f * 32 + half * 16);
    *dst = *src;
  }
  __syncthreads();

  const int rowA = wid * 16 + colLocal;
  const int col  = c0 + colLocal;
  const float bir = bi[col], biz = bi[DH + col], bin = bi[2 * DH + col];
  const float bhnc = bhn[col];

  unsigned* bcount = bar;
  unsigned* bgen   = bar + 1;
  const int s = laneHigh * 8;

  // per-lane LDS base for chunk 0; advances by 2*FBLK per kchunk
  const unsigned char* sBase =
      smem_raw + (size_t)colLocal * CSTRIDE + (size_t)laneHigh * FBLK;

  const f32x8 vzero = {0.f, 0.f, 0.f, 0.f, 0.f, 0.f, 0.f, 0.f};

#pragma unroll 1
  for (int t = 0; t < T_STEPS; ++t) {
    const unsigned short* xr = xbf + ((size_t)t * BATCH + rowA) * DH;
    const bool hzero = (t == 0) || (resets[t * BATCH + rowA] != 0);
    const unsigned short* hsrc =
        hzero ? zbf
              : hbuf + (size_t)(1 - (t & 1)) * BATCH * DH + (size_t)rowA * DH;
    unsigned short* hcur = hbuf + (size_t)(t & 1) * BATCH * DH;

    f32x8 acc[6];
#pragma unroll
    for (int i = 0; i < 6; ++i) acc[i] = vzero;

    // software-pipelined, double-buffered over 16 double-chunks of K=64
    FragAB f0[6], f1[6], a0x, a0h, a1x, a1h;
    const unsigned char* pB = sBase;
    loadB6(f0, pB);
    loadA(a0x, xr, 0, s);
    loadA(a0h, hsrc, 0, s);

#pragma unroll 1
    for (int i = 0; i < 16; ++i) {
      const int kc = i * 64;
      const int k1 = kc + 32;
      const int k2 = (kc + 64) & (DH - 1);   // wraps to 0 on last iteration
      const unsigned char* pBn = (i == 15) ? sBase : (pB + 4 * FBLK);

      // issue next-half loads first, then consume current set
      loadB6(f1, pB + 2 * FBLK);
      loadA(a1x, xr, k1, s);
      loadA(a1h, hsrc, k1, s);
      if (t + 1 < T_STEPS)
        __builtin_prefetch(xr + (size_t)BATCH * DH + kc, 0, 1);
      wmma6(a0x, a0h, f0, acc);

      loadB6(f0, pBn);
      loadA(a0x, xr, k2, s);
      loadA(a0h, hsrc, k2, s);
      wmma6(a1x, a1h, f1, acc);

      pB = pBn;
    }

    // one b64 load covers the 8 consecutive reset bytes this lane needs
    const int mBase = wid * 16 + laneHigh * 8;
    uint2 rpair = *(const uint2*)(resets + (size_t)t * BATCH + mBase);

#pragma unroll
    for (int r = 0; r < 8; ++r) {
      const int m = mBase + r;
      unsigned rbyte = (r < 4) ? ((rpair.x >> (8 * r)) & 0xFFu)
                               : ((rpair.y >> (8 * (r - 4))) & 0xFFu);
      bool zr = (t == 0) || (rbyte != 0u);
      const float* hp = zr ? zf32 : (out + ((size_t)(t - 1) * BATCH + m) * DH);
      float hold = hp[zr ? colLocal : col];
      float ir  = acc[0][r] + bir;
      float iz  = acc[1][r] + biz;
      float inn = acc[2][r] + bin;
      float hrv = acc[3][r], hzv = acc[4][r], hnv = acc[5][r];
      float rg = fsig(ir + hrv);
      float zg = fsig(iz + hzv);
      float ng = ftanh(inn + rg * (hnv + bhnc));
      float hnew = (1.0f - zg) * ng + zg * hold;
      out[((size_t)t * BATCH + m) * DH + col] = hnew;   // y_t (f32)
      hcur[(size_t)m * DH + col] = f2bf(hnew);          // carry (bf16)
    }

    grid_barrier(bcount, bgen, (unsigned)(t + 1));
  }
}

// ---------------- host launcher ----------------
extern "C" void kernel_launch(void* const* d_in, const int* in_sizes, int n_in,
                              void* d_out, int out_size, void* d_ws, size_t ws_size,
                              hipStream_t stream) {
  (void)in_sizes; (void)n_in; (void)out_size; (void)ws_size;
  const float* x   = (const float*)d_in[0];
  const float* Wi  = (const float*)d_in[1];
  const float* bi  = (const float*)d_in[2];
  const float* Wh  = (const float*)d_in[3];
  const float* bhn = (const float*)d_in[4];
  const unsigned char* resets = (const unsigned char*)d_in[5];
  float* out = (float*)d_out;

  const size_t WELEMS = (size_t)G3 * DH * DH;
  const size_t XELEMS = (size_t)T_STEPS * BATCH * DH;
  unsigned short* wiT  = (unsigned short*)d_ws;
  unsigned short* whT  = wiT + WELEMS;
  unsigned short* xbf  = whT + WELEMS;
  unsigned short* hbuf = xbf + XELEMS;
  unsigned short* zbf  = hbuf + (size_t)2 * BATCH * DH;
  float* zf32 = (float*)(zbf + DH);
  unsigned* bar = (unsigned*)(zf32 + DH);

  prep_weights<<<(unsigned)((WELEMS + 255) / 256), 256, 0, stream>>>(Wi, Wh, wiT,
                                                                     whT);
  prep_x<<<(unsigned)((XELEMS / 2 + 255) / 256), 256, 0, stream>>>(
      x, (unsigned*)xbf);
  init_misc<<<1, 1024, 0, stream>>>(bar, (unsigned*)zbf, zf32);

  hipFuncSetAttribute((const void*)gru_scan_kernel,
                      hipFuncAttributeMaxDynamicSharedMemorySize, (int)SMEMSZ);
  gru_scan_kernel<<<NBLK, NTHR, SMEMSZ, stream>>>(xbf, bi, bhn, resets, wiT, whT,
                                                  hbuf, zbf, zf32, bar, out);
}
